// ClassicSDE_79637283603160
// MI455X (gfx1250) — matrix operand order repaired and checked by reading, not compiled, not run on hardware
//
#include <hip/hip_runtime.h>
#include <hip/hip_bf16.h>
#include <stdint.h>

typedef __bf16 bf16;
typedef __attribute__((ext_vector_type(16))) __bf16 v16bf;
typedef __attribute__((ext_vector_type(8)))  __bf16 v8bf;
typedef __attribute__((ext_vector_type(8)))  float  v8f;

#define SDE_LEN    128
#define SDE_NOISE  16

// ---------------- helpers ----------------

__device__ __forceinline__ bf16 f2bf(float f) {
    return (bf16)f;   // native f32->bf16 convert (RNE)
}

__device__ __forceinline__ float lipswish(float v) {
    // 0.909 * v * sigmoid(v), sigmoid via fast v_rcp_f32
    float e = __expf(-v);
    return 0.909f * v * __builtin_amdgcn_rcpf(1.0f + e);
}

// Threefry-2x32, 20 rounds (JAX PRNG core)
__device__ __forceinline__ void tf2x32(uint32_t k0, uint32_t k1,
                                       uint32_t x0, uint32_t x1,
                                       uint32_t& o0, uint32_t& o1) {
    uint32_t k2 = 0x1BD11BDAu ^ k0 ^ k1;
    uint32_t v0 = x0 + k0, v1 = x1 + k1;
#define TF_R(r) { v0 += v1; v1 = (v1 << r) | (v1 >> (32 - r)); v1 ^= v0; }
    TF_R(13) TF_R(15) TF_R(26) TF_R(6)
    v0 += k1; v1 += k2 + 1u;
    TF_R(17) TF_R(29) TF_R(16) TF_R(24)
    v0 += k2; v1 += k0 + 2u;
    TF_R(13) TF_R(15) TF_R(26) TF_R(6)
    v0 += k0; v1 += k1 + 3u;
    TF_R(17) TF_R(29) TF_R(16) TF_R(24)
    v0 += k1; v1 += k2 + 4u;
    TF_R(13) TF_R(15) TF_R(26) TF_R(6)
    v0 += k2; v1 += k0 + 5u;
#undef TF_R
    o0 = v0; o1 = v1;
}

// single-precision inverse-erf (Giles 2010)
__device__ __forceinline__ float erfinv_f(float x) {
    float w = -__logf((1.0f - x) * (1.0f + x));
    float p;
    if (w < 5.0f) {
        w -= 2.5f;
        p = 2.81022636e-08f;
        p = 3.43273939e-07f + p * w;
        p = -3.5233877e-06f + p * w;
        p = -4.39150654e-06f + p * w;
        p = 0.00021858087f + p * w;
        p = -0.00125372503f + p * w;
        p = -0.00417768164f + p * w;
        p = 0.246640727f + p * w;
        p = 1.50140941f + p * w;
    } else {
        w = sqrtf(w) - 3.0f;
        p = -0.000200214257f;
        p = 0.000100950558f + p * w;
        p = 0.00134934322f + p * w;
        p = -0.00367342844f + p * w;
        p = 0.00573950773f + p * w;
        p = -0.0076224613f + p * w;
        p = 0.00943887047f + p * w;
        p = 1.00167406f + p * w;
        p = 2.83297682f + p * w;
    }
    return p * x;
}

// A fragment: 16x32 bf16 from LDS row-major (stride As elements).
__device__ __forceinline__ v16bf load_fragA(const bf16* A, int As, int kb,
                                            int laneN, int laneH) {
    const bf16* p = A + laneN * As + kb + laneH * 8;
    v8bf lo = *(const v8bf*)(p);
    v8bf hi = *(const v8bf*)(p + 16);
    v16bf r;
#pragma unroll
    for (int e = 0; e < 8; ++e) { r[e] = lo[e]; r[e + 8] = hi[e]; }
    return r;
}

// B fragment: 32x16 bf16 from global W^T (N rows x Ktot cols, row-major).
__device__ __forceinline__ v16bf load_fragB(const bf16* Wt, int Ktot, int n0,
                                            int kb, int laneN, int laneH) {
    const bf16* p = Wt + (size_t)(n0 + laneN) * Ktot + kb + laneH * 8;
    v8bf lo = *(const v8bf*)(p);
    v8bf hi = *(const v8bf*)(p + 16);
    v16bf r;
#pragma unroll
    for (int e = 0; e < 8; ++e) { r[e] = lo[e]; r[e + 8] = hi[e]; }
    return r;
}

// one 16x16 f32 output tile
__device__ __forceinline__ v8f gemm16(const bf16* A, int As, const bf16* Wt,
                                      int Ktot, int n0, int laneN, int laneH) {
    v8f acc = {0.f, 0.f, 0.f, 0.f, 0.f, 0.f, 0.f, 0.f};
    for (int kb = 0; kb < Ktot; kb += 32) {
        v16bf a = load_fragA(A, As, kb, laneN, laneH);
        v16bf b = load_fragB(Wt, Ktot, n0, kb, laneN, laneH);
        acc = __builtin_amdgcn_wmma_f32_16x16x32_bf16(
            false, a, false, b, (short)0, acc, false, false);
    }
    return acc;
}

// two 16x16 tiles sharing one A fragment per K-step (halves LDS A traffic,
// gives two independent WMMAs per A load)
__device__ __forceinline__ void gemm16x2(const bf16* A, int As, const bf16* Wt,
                                         int Ktot, int n0a, int n0b,
                                         int laneN, int laneH,
                                         v8f& acc0, v8f& acc1) {
    acc0 = (v8f){0.f, 0.f, 0.f, 0.f, 0.f, 0.f, 0.f, 0.f};
    acc1 = (v8f){0.f, 0.f, 0.f, 0.f, 0.f, 0.f, 0.f, 0.f};
    for (int kb = 0; kb < Ktot; kb += 32) {
        v16bf a  = load_fragA(A, As, kb, laneN, laneH);
        v16bf b0 = load_fragB(Wt, Ktot, n0a, kb, laneN, laneH);
        v16bf b1 = load_fragB(Wt, Ktot, n0b, kb, laneN, laneH);
        acc0 = __builtin_amdgcn_wmma_f32_16x16x32_bf16(
            false, a, false, b0, (short)0, acc0, false, false);
        acc1 = __builtin_amdgcn_wmma_f32_16x16x32_bf16(
            false, a, false, b1, (short)0, acc1, false, false);
    }
}

// hidden layer with Ntot = 256: each wave owns 2 tiles, shared-A path.
__device__ __forceinline__ void mlp_layer256(const bf16* A, int As, const bf16* Wt,
                                             const float* bias, int Ktot,
                                             bf16* Out, int Os) {
    int tid = threadIdx.x, lane = tid & 31, wave = tid >> 5;
    int laneN = lane & 15, laneH = lane >> 4;
    int n0 = wave * 32;
    v8f acc0, acc1;
    gemm16x2(A, As, Wt, Ktot, n0, n0 + 16, laneN, laneH, acc0, acc1);
    float bv0 = bias[n0 + laneN];
    float bv1 = bias[n0 + 16 + laneN];
#pragma unroll
    for (int r = 0; r < 8; ++r) {
        int M = r + 8 * laneH;
        Out[M * Os + n0 + laneN]      = f2bf(lipswish(acc0[r] + bv0));
        Out[M * Os + n0 + 16 + laneN] = f2bf(lipswish(acc1[r] + bv1));
    }
}

// ---------------- weight prep: transpose f32 (KxN) -> bf16 W^T (NxK) ----------------

__global__ void transpose_to_bf16(const float* __restrict__ src,
                                  bf16* __restrict__ dst, int K, int N) {
    int i = blockIdx.x * blockDim.x + threadIdx.x;
    if (i < K * N) {
        int n = i / K;
        int k = i - n * K;
        dst[i] = f2bf(src[(size_t)k * N + n]);
    }
}

// ---------------- persistent SDE kernel ----------------

struct SdeArgs {
    const float* ts;
    const float* initial;
    const bf16 *riW0, *riW1, *riW2, *riW3;
    const float *rib0, *rib1, *rib2, *rib3;
    const bf16 *drW0, *drW1, *drW2, *drW3;
    const float *drb0, *drb1, *drb2, *drb3, *drT;
    const bf16 *dfW0, *dfW1, *dfW2, *dfW3;
    const float *dfb0, *dfb1, *dfb2, *dfb3, *dfT;
    const bf16* roW;
    const float* rob;
    float* out;
};

__global__ __launch_bounds__(256) void sde_scan_kernel(SdeArgs A) {
    __shared__ __attribute__((aligned(16))) bf16 sH[16 * 136];
    __shared__ __attribute__((aligned(16))) bf16 sB1[16 * 264];
    __shared__ __attribute__((aligned(16))) bf16 sB2[16 * 264];
    __shared__ __attribute__((aligned(16))) float sX[16 * 132];
    __shared__ float sBeffDr[256];
    __shared__ float sBeffDf[256];
    __shared__ float sBM[16 * 16];

    const int tid = threadIdx.x;
    const int lane = tid & 31, wave = tid >> 5;
    const int laneN = lane & 15, laneH = lane >> 4;
    const int b0 = blockIdx.x * 16;

    // ---- prolog: x0 = read_in MLP(initial) ----
    for (int idx = tid; idx < 16 * 64; idx += 256) {
        int m = idx >> 6, c = idx & 63;
        sH[m * 136 + c] = f2bf(A.initial[(size_t)(b0 + m) * 64 + c]);
    }
    __syncthreads();
    mlp_layer256(sH, 136, A.riW0, A.rib0, 64, sB1, 264);
    __syncthreads();
    mlp_layer256(sB1, 264, A.riW1, A.rib1, 256, sB2, 264);
    __syncthreads();
    mlp_layer256(sB2, 264, A.riW2, A.rib2, 256, sB1, 264);
    __syncthreads();
    {
        int n0 = wave * 16;
        v8f acc = gemm16(sB1, 264, A.riW3, 256, n0, laneN, laneH);
        float bv = A.rib3[n0 + laneN];
#pragma unroll
        for (int r = 0; r < 8; ++r)
            sX[(r + 8 * laneH) * 132 + n0 + laneN] = acc[r] + bv;
    }
    __syncthreads();

    // ---- time scan ----
    for (int it = 0; it < SDE_LEN; ++it) {
        for (int idx = tid; idx < 16 * 128; idx += 256) {
            int m = idx >> 7, c = idx & 127;
            sH[m * 136 + c] = f2bf(sX[m * 132 + c]);
        }
        float tcur = A.ts[it];
        float dtv = (it < SDE_LEN - 1) ? (A.ts[it + 1] - tcur) : 0.0f;
        if (it < SDE_LEN - 1) {
            sBeffDr[tid] = A.drb0[tid] + tcur * A.drT[tid];
            sBeffDf[tid] = A.dfb0[tid] + tcur * A.dfT[tid];
            uint32_t kk0, kk1;
            tf2x32(0u, 42u, 0u, (uint32_t)it, kk0, kk1);
            int row = tid >> 4, nn = tid & 15;
            uint32_t flat = (uint32_t)((b0 + row) * SDE_NOISE + nn);
            uint32_t pos = (flat < 4096u) ? flat : flat - 4096u;
            uint32_t o0, o1;
            tf2x32(kk0, kk1, pos, pos + 4096u, o0, o1);
            uint32_t bits = (flat < 4096u) ? o0 : o1;
            float f01 = (float)(bits >> 8) * (1.0f / 16777216.0f);
            float u = 2.0f * f01 - 1.0f;
            u = fminf(0.9999997f, fmaxf(-0.9999997f, u));
            sBM[tid] = 1.41421356f * erfinv_f(u) * sqrtf(dtv);
        }
        __syncthreads();

        // readout: out[b][it][1:] = x @ Wro + bro ; out[b][it][0] = ts[it]
        if (wave < 4) {
            int n0 = wave * 16;
            v8f acc = gemm16(sH, 136, A.roW, 128, n0, laneN, laneH);
            float bv = A.rob[n0 + laneN];
#pragma unroll
            for (int r = 0; r < 8; ++r) {
                int M = r + 8 * laneH;
                A.out[((size_t)(b0 + M) * SDE_LEN + it) * 65 + 1 + n0 + laneN] =
                    acc[r] + bv;
            }
        } else if (wave == 4) {
            if (lane < 16)
                A.out[((size_t)(b0 + lane) * SDE_LEN + it) * 65] = tcur;
        }
        if (it == SDE_LEN - 1) break;

        // ---- drift MLP ----
        mlp_layer256(sH, 136, A.drW0, sBeffDr, 128, sB1, 264);
        __syncthreads();
        mlp_layer256(sB1, 264, A.drW1, A.drb1, 256, sB2, 264);
        __syncthreads();
        mlp_layer256(sB2, 264, A.drW2, A.drb2, 256, sB1, 264);
        __syncthreads();
        {
            int n0 = wave * 16;
            v8f acc = gemm16(sB1, 264, A.drW3, 256, n0, laneN, laneH);
            float bv = A.drb3[n0 + laneN];
#pragma unroll
            for (int r = 0; r < 8; ++r) {
                int M = r + 8 * laneH;
                sX[M * 132 + n0 + laneN] += (acc[r] + bv) * dtv;
            }
        }
        __syncthreads();

        // ---- diffusion MLP ----
        mlp_layer256(sH, 136, A.dfW0, sBeffDf, 128, sB1, 264);
        __syncthreads();
        mlp_layer256(sB1, 264, A.dfW1, A.dfb1, 256, sB2, 264);
        __syncthreads();
        mlp_layer256(sB2, 264, A.dfW2, A.dfb2, 256, sB1, 264);
        __syncthreads();
        {   // final N=2048 layer fused with einsum('bhn,bn->bh'); tiles in pairs
            float bm8[8];
#pragma unroll
            for (int r = 0; r < 8; ++r)
                bm8[r] = sBM[(r + 8 * laneH) * 16 + laneN];
            for (int t = 0; t < 16; t += 2) {
                int n0 = wave * 256 + t * 16;
                v8f acc0, acc1;
                gemm16x2(sB1, 264, A.dfW3, 256, n0, n0 + 16, laneN, laneH,
                         acc0, acc1);
                float bv0 = A.dfb3[n0 + laneN];
                float bv1 = A.dfb3[n0 + 16 + laneN];
                float s0[8], s1[8];
#pragma unroll
                for (int r = 0; r < 8; ++r) {
                    s0[r] = (acc0[r] + bv0) * bm8[r];
                    s1[r] = (acc1[r] + bv1) * bm8[r];
                }
#pragma unroll
                for (int m = 1; m < 16; m <<= 1) {
#pragma unroll
                    for (int r = 0; r < 8; ++r) {
                        s0[r] += __shfl_xor(s0[r], m, 32);
                        s1[r] += __shfl_xor(s1[r], m, 32);
                    }
                }
                if (laneN == 0) {
                    int h0 = wave * 16 + t;
#pragma unroll
                    for (int r = 0; r < 8; ++r) {
                        sX[(r + 8 * laneH) * 132 + h0]     += s0[r];
                        sX[(r + 8 * laneH) * 132 + h0 + 1] += s1[r];
                    }
                }
            }
        }
        __syncthreads();
    }
}

// ---------------- host ----------------

extern "C" void kernel_launch(void* const* d_in, const int* in_sizes, int n_in,
                              void* d_out, int out_size, void* d_ws, size_t ws_size,
                              hipStream_t stream) {
    (void)in_sizes; (void)n_in; (void)out_size; (void)ws_size;

    const float* ts      = (const float*)d_in[0];
    const float* initial = (const float*)d_in[3];

    const float* riWf[4] = {(const float*)d_in[4], (const float*)d_in[6],
                            (const float*)d_in[8], (const float*)d_in[10]};
    const float* rib[4]  = {(const float*)d_in[5], (const float*)d_in[7],
                            (const float*)d_in[9], (const float*)d_in[11]};
    const float* roWf = (const float*)d_in[12];
    const float* rob  = (const float*)d_in[13];
    const float* drWf[4] = {(const float*)d_in[14], (const float*)d_in[16],
                            (const float*)d_in[18], (const float*)d_in[20]};
    const float* drb[4]  = {(const float*)d_in[15], (const float*)d_in[17],
                            (const float*)d_in[19], (const float*)d_in[21]};
    const float* dfWf[4] = {(const float*)d_in[22], (const float*)d_in[24],
                            (const float*)d_in[26], (const float*)d_in[28]};
    const float* dfb[4]  = {(const float*)d_in[23], (const float*)d_in[25],
                            (const float*)d_in[27], (const float*)d_in[29]};

    bf16* base = (bf16*)d_ws;
    size_t off = 0;
    auto nxt = [&](size_t n) { bf16* p = base + off; off += n; return p; };
    bf16* riW0 = nxt(256 * 64);
    bf16* riW1 = nxt(256 * 256);
    bf16* riW2 = nxt(256 * 256);
    bf16* riW3 = nxt(128 * 256);
    bf16* drW0 = nxt(256 * 128);
    bf16* drW1 = nxt(256 * 256);
    bf16* drW2 = nxt(256 * 256);
    bf16* drW3 = nxt(128 * 256);
    bf16* dfW0 = nxt(256 * 128);
    bf16* dfW1 = nxt(256 * 256);
    bf16* dfW2 = nxt(256 * 256);
    bf16* dfW3 = nxt(2048 * 256);
    bf16* roWt = nxt(64 * 128);

    auto tr = [&](const float* src, bf16* dst, int K, int N) {
        int tot = K * N;
        transpose_to_bf16<<<(tot + 255) / 256, 256, 0, stream>>>(src, dst, K, N);
    };
    tr(riWf[0], riW0, 64, 256);
    tr(riWf[1], riW1, 256, 256);
    tr(riWf[2], riW2, 256, 256);
    tr(riWf[3], riW3, 256, 128);
    tr(drWf[0], drW0, 128, 256);
    tr(drWf[1], drW1, 256, 256);
    tr(drWf[2], drW2, 256, 256);
    tr(drWf[3], drW3, 256, 128);
    tr(dfWf[0], dfW0, 128, 256);
    tr(dfWf[1], dfW1, 256, 256);
    tr(dfWf[2], dfW2, 256, 256);
    tr(dfWf[3], dfW3, 256, 2048);
    tr(roWf, roWt, 128, 64);

    SdeArgs a;
    a.ts = ts; a.initial = initial;
    a.riW0 = riW0; a.riW1 = riW1; a.riW2 = riW2; a.riW3 = riW3;
    a.rib0 = rib[0]; a.rib1 = rib[1]; a.rib2 = rib[2]; a.rib3 = rib[3];
    a.drW0 = drW0; a.drW1 = drW1; a.drW2 = drW2; a.drW3 = drW3;
    a.drb0 = drb[0]; a.drb1 = drb[1]; a.drb2 = drb[2]; a.drb3 = drb[3];
    a.drT = drWf[0] + 128 * 256;
    a.dfW0 = dfW0; a.dfW1 = dfW1; a.dfW2 = dfW2; a.dfW3 = dfW3;
    a.dfb0 = dfb[0]; a.dfb1 = dfb[1]; a.dfb2 = dfb[2]; a.dfb3 = dfb[3];
    a.dfT = dfWf[0] + 128 * 256;
    a.roW = roWt; a.rob = rob;
    a.out = (float*)d_out;

    sde_scan_kernel<<<32, 256, 0, stream>>>(a);
}
